// Attention_60885456388867
// MI455X (gfx1250) — matrix operand order repaired and checked by reading, not compiled, not run on hardware
//
#include <hip/hip_runtime.h>
#include <math.h>

typedef __attribute__((ext_vector_type(16))) _Float16 v16h;
typedef __attribute__((ext_vector_type(8)))  _Float16 v8h;
typedef __attribute__((ext_vector_type(2)))  _Float16 h2;
typedef __attribute__((ext_vector_type(2)))  __fp16   fp16x2;
typedef __attribute__((ext_vector_type(8)))  float    v8f;

#define NB   64
#define LSEQ 1024
#define DD   64
#define EPSN 1e-5f
// (1/sqrt(32)) * log2(e): scores come out of WMMA already in exp2 domain
#define QS2  0.25501653761406407f

__device__ __forceinline__ h2 pk2(float a, float b) {
  fp16x2 t = __builtin_amdgcn_cvt_pkrtz(a, b);   // v_cvt_pk_rtz_f16_f32
  return __builtin_bit_cast(h2, t);
}

__device__ __forceinline__ v8f wmma16(v16h a, v16h b, v8f c) {
  return __builtin_amdgcn_wmma_f32_16x16x32_f16(false, a, false, b,
                                                (short)0, c, false, false);
}

// Load a 16x32 f16 A-operand tile from an f32 row-major weight matrix.
// Rows = mrow (per-lane M index), K-dims = koff + {16h..16h+15}.
__device__ __forceinline__ v16h load_wA(const float* __restrict__ w, int ncols,
                                        int mrow, int koff, int h) {
  const float* p = w + (size_t)mrow * ncols + koff + h * 16;
  float4 x0 = reinterpret_cast<const float4*>(p)[0];
  float4 x1 = reinterpret_cast<const float4*>(p)[1];
  float4 y0 = reinterpret_cast<const float4*>(p + 8)[0];
  float4 y1 = reinterpret_cast<const float4*>(p + 8)[1];
  v16h A;
  h2* ap = reinterpret_cast<h2*>(&A);
  ap[0] = pk2(x0.x, x0.y);
  ap[1] = pk2(x0.z, x0.w);
  ap[2] = pk2(x1.x, x1.y);
  ap[3] = pk2(x1.z, x1.w);
  ap[4] = pk2(y0.x, y0.y);
  ap[5] = pk2(y0.z, y0.w);
  ap[6] = pk2(y1.x, y1.y);
  ap[7] = pk2(y1.z, y1.w);
  return A;
}

// C accumulator initialized with a broadcast bias: C[m][*] = bias[16t+8h+m]
__device__ __forceinline__ v8f bias_c8(const float* __restrict__ bias, int t, int h) {
  v8f c;
  #pragma unroll
  for (int r = 0; r < 8; r++) c[r] = bias[16 * t + 8 * h + r];
  return c;
}

// Build the two K-chunk B-operands from a full 64-f32 row held in registers.
// B layout: lane(h,n) element i = row[32c + 16h + i].
__device__ __forceinline__ void build_B(const float* row, int h, v16h* B0, v16h* B1) {
  h2* b0 = reinterpret_cast<h2*>(B0);
  h2* b1 = reinterpret_cast<h2*>(B1);
  #pragma unroll
  for (int j = 0; j < 8; j++) {
    float a0 = h ? row[16 + 2*j]     : row[2*j];
    float a1 = h ? row[16 + 2*j + 1] : row[2*j + 1];
    b0[j] = pk2(a0, a1);
    float c0 = h ? row[48 + 2*j]     : row[32 + 2*j];
    float c1 = h ? row[48 + 2*j + 1] : row[32 + 2*j + 1];
    b1[j] = pk2(c0, c1);
  }
}

// Full-row load + layernorm into out[64] (registers).
__device__ __forceinline__ void load_ln(const float* __restrict__ src,
                                        const float* __restrict__ w,
                                        const float* __restrict__ bb,
                                        float* out) {
  float r[DD];
  float mu = 0.f;
  #pragma unroll
  for (int i = 0; i < DD / 4; i++) {
    float4 t = reinterpret_cast<const float4*>(src)[i];
    r[4*i+0] = t.x; r[4*i+1] = t.y; r[4*i+2] = t.z; r[4*i+3] = t.w;
    mu += t.x + t.y + t.z + t.w;
  }
  mu *= (1.f / DD);
  float var = 0.f;
  #pragma unroll
  for (int i = 0; i < DD; i++) { float d = r[i] - mu; var += d * d; }
  var *= (1.f / DD);
  float rs = rsqrtf(var + EPSN);
  #pragma unroll
  for (int i = 0; i < DD; i++) out[i] = (r[i] - mu) * rs * w[i] + bb[i];
}

// Store one 32-f32 half of a row (h selects which half, constant indices).
__device__ __forceinline__ void store_half_row(float* dst, const float* row, int h) {
  if (h) {
    #pragma unroll
    for (int i = 0; i < 8; i++) {
      float4 t; t.x = row[32+4*i]; t.y = row[33+4*i]; t.z = row[34+4*i]; t.w = row[35+4*i];
      reinterpret_cast<float4*>(dst + 32)[i] = t;
    }
  } else {
    #pragma unroll
    for (int i = 0; i < 8; i++) {
      float4 t; t.x = row[4*i]; t.y = row[1+4*i]; t.z = row[2+4*i]; t.w = row[3+4*i];
      reinterpret_cast<float4*>(dst)[i] = t;
    }
  }
}

// ---------------------------------------------------------------------------
// Kernel 1: LN(y) -> yn (output #2); k = yn@k_w^T+k_b (f16 [row][32]);
//           vT[b][d][row] = yn@v_w^T+v_b (f16, transposed).  One wave / 16 rows.
// ---------------------------------------------------------------------------
__global__ __launch_bounds__(128) void kern_y(
    const float* __restrict__ y,
    const float* __restrict__ ln2_w, const float* __restrict__ ln2_b,
    const float* __restrict__ k_w,   const float* __restrict__ k_b,
    const float* __restrict__ v_w,   const float* __restrict__ v_b,
    float* __restrict__ yn_out, _Float16* __restrict__ kf, _Float16* __restrict__ vT)
{
  int wid  = blockIdx.x * 4 + (threadIdx.x >> 5);
  int lane = threadIdx.x & 31;
  int h = lane >> 4, n = lane & 15;
  int rg = wid * 16 + n;                 // this lane's row (dup across halves)
  int b = rg >> 10, row = rg & 1023;

  float yn[DD];
  load_ln(y + (size_t)rg * DD, ln2_w, ln2_b, yn);
  store_half_row(yn_out + (size_t)rg * DD, yn, h);

  v16h B0, B1;
  build_B(yn, h, &B0, &B1);

  // k projection: 2 N-tiles x 2 K-chunks
  #pragma unroll
  for (int t = 0; t < 2; t++) {
    v8f c = bias_c8(k_b, t, h);
    c = wmma16(load_wA(k_w, DD, 16*t + n, 0,  h), B0, c);
    c = wmma16(load_wA(k_w, DD, 16*t + n, 32, h), B1, c);
    v8h ks;
    #pragma unroll
    for (int r = 0; r < 8; r++) ks[r] = (_Float16)c[r];
    *reinterpret_cast<v8h*>(kf + (size_t)rg * 32 + 16*t + 8*h) = ks;
  }
  // v projection: 4 N-tiles x 2 K-chunks, stored transposed
  #pragma unroll
  for (int t = 0; t < 4; t++) {
    v8f c = bias_c8(v_b, t, h);
    c = wmma16(load_wA(v_w, DD, 16*t + n, 0,  h), B0, c);
    c = wmma16(load_wA(v_w, DD, 16*t + n, 32, h), B1, c);
    #pragma unroll
    for (int r = 0; r < 8; r++) {
      int outd = 16*t + 8*h + r;
      vT[(((size_t)b * DD + outd) << 10) + row] = (_Float16)c[r];
    }
  }
}

// ---------------------------------------------------------------------------
// Kernel 2: LN(x) -> xn (f32 ws); q = (xn@q_w^T+q_b) * log2e/sqrt(32) (f16 ws)
// ---------------------------------------------------------------------------
__global__ __launch_bounds__(128) void kern_x(
    const float* __restrict__ x,
    const float* __restrict__ ln1_w, const float* __restrict__ ln1_b,
    const float* __restrict__ q_w,   const float* __restrict__ q_b,
    _Float16* __restrict__ qf, float* __restrict__ xnh)
{
  int wid  = blockIdx.x * 4 + (threadIdx.x >> 5);
  int lane = threadIdx.x & 31;
  int h = lane >> 4, n = lane & 15;
  int rg = wid * 16 + n;

  float xn[DD];
  load_ln(x + (size_t)rg * DD, ln1_w, ln1_b, xn);
  store_half_row(xnh + (size_t)rg * DD, xn, h);

  v16h B0, B1;
  build_B(xn, h, &B0, &B1);

  #pragma unroll
  for (int t = 0; t < 2; t++) {
    v8f c = bias_c8(q_b, t, h);
    c = wmma16(load_wA(q_w, DD, 16*t + n, 0,  h), B0, c);
    c = wmma16(load_wA(q_w, DD, 16*t + n, 32, h), B1, c);
    v8h qs;
    #pragma unroll
    for (int r = 0; r < 8; r++) qs[r] = (_Float16)(c[r] * QS2);
    *reinterpret_cast<v8h*>(qf + (size_t)rg * 32 + 16*t + 8*h) = qs;
  }
}

// ---------------------------------------------------------------------------
// Kernel 3: flash attention, one wave per 16-row q tile (exp2 domain).
// ---------------------------------------------------------------------------
__global__ __launch_bounds__(128) void kern_attn(
    const _Float16* __restrict__ qf, const _Float16* __restrict__ kf,
    const _Float16* __restrict__ vT, float* __restrict__ xnh)
{
  const int wave = threadIdx.x >> 5;
  const int lane = threadIdx.x & 31;
  const int tile = blockIdx.x * 4 + wave;
  const int b  = tile >> 6;
  const int q0 = (tile & 63) << 4;
  const int h  = lane >> 4;
  const int n  = lane & 15;

  v16h qb;
  {
    const _Float16* qr = qf + (((size_t)(b << 10) + q0 + n) << 5) + (h << 3);
    v8h lo = *reinterpret_cast<const v8h*>(qr);
    v8h hi = *reinterpret_cast<const v8h*>(qr + 16);
    #pragma unroll
    for (int i = 0; i < 8; i++) { qb[i] = lo[i]; qb[8 + i] = hi[i]; }
  }

  const _Float16* kbase = kf + ((size_t)(b << 10) << 5);
  const _Float16* vbase = vT + ((size_t)b * DD << 10);

  v8f acc[4];
  #pragma unroll
  for (int t = 0; t < 4; t++)
    #pragma unroll
    for (int rr = 0; rr < 8; rr++) acc[t][rr] = 0.f;
  v8f zero8;
  #pragma unroll
  for (int rr = 0; rr < 8; rr++) zero8[rr] = 0.f;

  float mrun = -1e30f, lsum = 0.f;

  for (int c = 0; c < LSEQ; c += 32) {
    v16h ka0, ka1;
    {
      const _Float16* kr = kbase + ((size_t)(c + n) << 5) + (h << 3);
      v8h lo = *reinterpret_cast<const v8h*>(kr);
      v8h hi = *reinterpret_cast<const v8h*>(kr + 16);
      #pragma unroll
      for (int i = 0; i < 8; i++) { ka0[i] = lo[i]; ka0[8 + i] = hi[i]; }
      kr += (size_t)16 << 5;
      lo = *reinterpret_cast<const v8h*>(kr);
      hi = *reinterpret_cast<const v8h*>(kr + 16);
      #pragma unroll
      for (int i = 0; i < 8; i++) { ka1[i] = lo[i]; ka1[8 + i] = hi[i]; }
    }
    if (c + 32 < LSEQ)
      __builtin_prefetch(kbase + ((size_t)(c + 32 + n) << 5), 0, 1);

    v8f s0 = wmma16(ka0, qb, zero8);   // scores already in log2 units
    v8f s1 = wmma16(ka1, qb, zero8);

    float cm = -1e30f;
    #pragma unroll
    for (int rr = 0; rr < 8; rr++) cm = fmaxf(cm, fmaxf(s0[rr], s1[rr]));
    cm = fmaxf(cm, __shfl_xor(cm, 16, 32));
    float mnew  = fmaxf(mrun, cm);
    float alpha = __builtin_amdgcn_exp2f(mrun - mnew);
    mrun = mnew;

    float p0[8], p1[8], cs = 0.f;
    #pragma unroll
    for (int rr = 0; rr < 8; rr++) {
      p0[rr] = __builtin_amdgcn_exp2f(s0[rr] - mnew);
      p1[rr] = __builtin_amdgcn_exp2f(s1[rr] - mnew);
      cs += p0[rr] + p1[rr];
    }
    cs += __shfl_xor(cs, 16, 32);
    lsum = lsum * alpha + cs;

    // pack P^T (C layout -> B layout): one cross-half exchange + pk convert
    v16h pb;
    h2* pp = reinterpret_cast<h2*>(&pb);
    #pragma unroll
    for (int j = 0; j < 8; j++) {
      int jj = (j < 4) ? 2 * j : 2 * j - 8;
      float send0 = h ? p0[jj]     : p1[jj];
      float send1 = h ? p0[jj + 1] : p1[jj + 1];
      float r0 = __shfl_xor(send0, 16, 32);
      float r1 = __shfl_xor(send1, 16, 32);
      float e0, e1;
      if (j < 4) { e0 = h ? r0 : p0[jj];     e1 = h ? r1 : p0[jj + 1]; }
      else       { e0 = h ? p1[jj] : r0;     e1 = h ? p1[jj + 1] : r1; }
      pp[j] = pk2(e0, e1);
    }

    #pragma unroll
    for (int t = 0; t < 4; t++)
      #pragma unroll
      for (int rr = 0; rr < 8; rr++) acc[t][rr] *= alpha;

    #pragma unroll
    for (int t = 0; t < 4; t++) {
      const _Float16* vr = vbase + ((size_t)(16 * t + n) << 10) + c + (h << 4);
      v8h lo = *reinterpret_cast<const v8h*>(vr);
      v8h hi = *reinterpret_cast<const v8h*>(vr + 8);
      v16h va;
      #pragma unroll
      for (int i = 0; i < 8; i++) { va[i] = lo[i]; va[8 + i] = hi[i]; }
      acc[t] = wmma16(va, pb, acc[t]);
    }
  }

  float linv = 1.f / lsum;
  #pragma unroll
  for (int t = 0; t < 4; t++) {
    #pragma unroll
    for (int rr = 0; rr < 8; rr++) {
      int d = 16 * t + rr + 8 * h;
      size_t idx = (((size_t)(b << 10) + q0 + n) << 6) + d;
      xnh[idx] = xnh[idx] + acc[t][rr] * linv;
    }
  }
}

// ---------------------------------------------------------------------------
// Kernel 4: hn = LN3(h); out1 = hn + relu(hn@in_w^T+in_b)@out_w^T + out_b
// GEMM1 computes t^T = in_w @ hn^T; GEMM2 computes mlp^T = out_w @ t^T with
// the same C->B layout repack as the attention P tile.
// ---------------------------------------------------------------------------
__global__ __launch_bounds__(128) void kern_mlp(
    const float* __restrict__ xnh,
    const float* __restrict__ ln3_w, const float* __restrict__ ln3_b,
    const float* __restrict__ in_w,  const float* __restrict__ in_b,
    const float* __restrict__ out_w, const float* __restrict__ out_b,
    float* __restrict__ out1)
{
  int wid  = blockIdx.x * 4 + (threadIdx.x >> 5);
  int lane = threadIdx.x & 31;
  int h = lane >> 4, n = lane & 15;
  int rg = wid * 16 + n;

  float hn[DD];
  load_ln(xnh + (size_t)rg * DD, ln3_w, ln3_b, hn);

  v16h B0, B1;
  build_B(hn, h, &B0, &B1);

  // GEMM1: t^T (32 x 16) = in_w @ hn^T, + bias, relu
  v8f t0 = bias_c8(in_b, 0, h);
  t0 = wmma16(load_wA(in_w, DD, n,      0,  h), B0, t0);
  t0 = wmma16(load_wA(in_w, DD, n,      32, h), B1, t0);
  v8f t1 = bias_c8(in_b, 1, h);
  t1 = wmma16(load_wA(in_w, DD, 16 + n, 0,  h), B0, t1);
  t1 = wmma16(load_wA(in_w, DD, 16 + n, 32, h), B1, t1);
  #pragma unroll
  for (int r = 0; r < 8; r++) { t0[r] = fmaxf(t0[r], 0.f); t1[r] = fmaxf(t1[r], 0.f); }

  // repack t^T (C layout) -> B operand (K=32 hidden, N=16 rows)
  v16h pb;
  h2* pp = reinterpret_cast<h2*>(&pb);
  #pragma unroll
  for (int j = 0; j < 8; j++) {
    int jj = (j < 4) ? 2 * j : 2 * j - 8;
    float send0 = h ? t0[jj]     : t1[jj];
    float send1 = h ? t0[jj + 1] : t1[jj + 1];
    float r0 = __shfl_xor(send0, 16, 32);
    float r1 = __shfl_xor(send1, 16, 32);
    float e0, e1;
    if (j < 4) { e0 = h ? r0 : t0[jj];     e1 = h ? r1 : t0[jj + 1]; }
    else       { e0 = h ? t1[jj] : r0;     e1 = h ? t1[jj + 1] : r1; }
    pp[j] = pk2(e0, e1);
  }

  // GEMM2: mlp^T (64 x 16) = out_w @ t^T, + out_b; out1 = hn + mlp
  #pragma unroll
  for (int t = 0; t < 4; t++) {
    v8f m_ = bias_c8(out_b, t, h);
    m_ = wmma16(load_wA(out_w, 32, 16*t + n, 0, h), pb, m_);
    float* base = out1 + (size_t)rg * DD + 16*t + 8*h;
    float4 o0, o1;
    o0.x = m_[0] + (h ? hn[16*t + 8]  : hn[16*t + 0]);
    o0.y = m_[1] + (h ? hn[16*t + 9]  : hn[16*t + 1]);
    o0.z = m_[2] + (h ? hn[16*t + 10] : hn[16*t + 2]);
    o0.w = m_[3] + (h ? hn[16*t + 11] : hn[16*t + 3]);
    o1.x = m_[4] + (h ? hn[16*t + 12] : hn[16*t + 4]);
    o1.y = m_[5] + (h ? hn[16*t + 13] : hn[16*t + 5]);
    o1.z = m_[6] + (h ? hn[16*t + 14] : hn[16*t + 6]);
    o1.w = m_[7] + (h ? hn[16*t + 15] : hn[16*t + 7]);
    reinterpret_cast<float4*>(base)[0] = o0;
    reinterpret_cast<float4*>(base)[1] = o1;
  }
}

// ---------------------------------------------------------------------------
extern "C" void kernel_launch(void* const* d_in, const int* in_sizes, int n_in,
                              void* d_out, int out_size, void* d_ws, size_t ws_size,
                              hipStream_t stream) {
  const float* x     = (const float*)d_in[0];
  const float* y     = (const float*)d_in[1];
  const float* ln1_w = (const float*)d_in[2];
  const float* ln1_b = (const float*)d_in[3];
  const float* ln2_w = (const float*)d_in[4];
  const float* ln2_b = (const float*)d_in[5];
  const float* ln3_w = (const float*)d_in[6];
  const float* ln3_b = (const float*)d_in[7];
  const float* q_w   = (const float*)d_in[8];
  const float* q_b   = (const float*)d_in[9];
  const float* k_w   = (const float*)d_in[10];
  const float* k_b   = (const float*)d_in[11];
  const float* v_w   = (const float*)d_in[12];
  const float* v_b   = (const float*)d_in[13];
  const float* in_w  = (const float*)d_in[14];
  const float* in_b  = (const float*)d_in[15];
  const float* out_w = (const float*)d_in[16];
  const float* out_b = (const float*)d_in[17];

  float* out1   = (float*)d_out;
  float* yn_out = out1 + (size_t)NB * LSEQ * DD;

  char* ws = (char*)d_ws;
  _Float16* qf  = (_Float16*)(ws);                       // 4 MiB
  _Float16* kf  = (_Float16*)(ws + ((size_t)4  << 20));  // 4 MiB
  _Float16* vTb = (_Float16*)(ws + ((size_t)8  << 20));  // 8 MiB
  float*    xnh = (float*)   (ws + ((size_t)16 << 20));  // 16 MiB (xn -> h in place)

  const int nblk = (NB * LSEQ / 16) / 4;  // 1024 blocks of 4 waves
  kern_y<<<nblk, 128, 0, stream>>>(y, ln2_w, ln2_b, k_w, k_b, v_w, v_b,
                                   yn_out, kf, vTb);
  kern_x<<<nblk, 128, 0, stream>>>(x, ln1_w, ln1_b, q_w, q_b, qf, xnh);
  kern_attn<<<(NB * 64) / 4, 128, 0, stream>>>(qf, kf, vTb, xnh);
  kern_mlp<<<nblk, 128, 0, stream>>>(xnh, ln3_w, ln3_b, in_w, in_b,
                                     out_w, out_b, out1);
}
